// GluedAttnProcessor_48232482734502
// MI455X (gfx1250) — matrix operand order repaired and checked by reading, not compiled, not run on hardware
//
#include <hip/hip_runtime.h>
#include <hip/hip_bf16.h>
#include <math.h>

typedef __attribute__((ext_vector_type(16))) _Float16 v16h;
typedef __attribute__((ext_vector_type(8)))  float    v8f;

#define HEADS 24
#define HDIM  128
#define BS    256
#define QBLK  54     // 3*3*6 query tile grid
#define KVBLK 120    // 3*5*8 kv tile grid

// ---------------------------------------------------------------------------
// RoPE: pair p in [0,64): p<8 -> t axis (dim 16), p<36 -> h axis (dim 56),
// else w axis (dim 56). theta=256 => freq = 2^(-16*i/dim).
// ---------------------------------------------------------------------------
__device__ __forceinline__ void rope_rotate(float pos, int p, float x0, float x1,
                                            float& y0, float& y1) {
  float i, d;
  if (p < 8)       { i = (float)p;        d = 16.0f; }
  else if (p < 36) { i = (float)(p - 8);  d = 56.0f; }
  else             { i = (float)(p - 36); d = 56.0f; }
  float freq = exp2f(-16.0f * i / d);
  float ang = pos * freq;
  float s = sinf(ang);
  float c = cosf(ang);
  y0 = x0 * c - x1 * s;
  y1 = x1 * c + x0 * s;
}

// Q preprocess: RoPE + tile -> Qh[head][qb][row][e] f16 (row-major)
__global__ __launch_bounds__(256) void rope_q_kernel(const float* __restrict__ q,
                                                     _Float16* __restrict__ Qh) {
  int gid = blockIdx.x * 256 + threadIdx.x;     // QBLK*BS*HEADS*64 threads
  int p = gid & 63;
  int head = (gid >> 6) % HEADS;
  int rowglob = gid / (64 * HEADS);
  int qb = rowglob >> 8;
  int row = rowglob & 255;
  int bt = qb / 18, bh = (qb / 6) % 3, bw = qb % 6;
  int tt = row >> 6, th = (row >> 3) & 7, tw = row & 7;
  int t = bt * 4 + tt, h = bh * 8 + th, w = bw * 8 + tw;
  int l = (t * 24 + h) * 48 + w;
  float pos = (p < 8) ? (float)t : (p < 36) ? (float)h : (float)w;
  const float2 x = *(const float2*)(q + ((size_t)l * HEADS + head) * HDIM + 2 * p);
  float y0, y1;
  rope_rotate(pos, p, x.x, x.y, y0, y1);
  union { _Float16 h2[2]; unsigned u; } pk;
  pk.h2[0] = (_Float16)y0;
  pk.h2[1] = (_Float16)y1;
  *(unsigned*)(Qh + (((size_t)(head * QBLK + qb) * BS + row) * HDIM + 2 * p)) = pk.u;
}

// K preprocess: glue (wrap) + RoPE with padded positions + tile
// -> Kh[head][kb][row][e] f16
__global__ __launch_bounds__(256) void rope_k_kernel(const float* __restrict__ k,
                                                     _Float16* __restrict__ Kh) {
  int gid = blockIdx.x * 256 + threadIdx.x;     // KVBLK*BS*HEADS*64 threads
  int p = gid & 63;
  int head = (gid >> 6) % HEADS;
  int rowglob = gid / (64 * HEADS);
  int kb = rowglob >> 8;
  int row = rowglob & 255;
  int kt = kb / 40, kh = (kb / 8) % 5, kw = kb % 8;
  int tt = row >> 6, th = (row >> 3) & 7, tw = row & 7;
  int t = kt * 4 + tt;             // 0..11, not glued
  int hp = kh * 8 + th;            // 0..39 padded
  int wp = kw * 8 + tw;            // 0..63 padded
  float pos = (p < 8) ? (float)t : (p < 36) ? (float)(hp - 8) : (float)(wp - 8);
  int h = (hp + 16) % 24;          // (hp-8) mod 24
  int w = (wp + 40) % 48;          // (wp-8) mod 48
  int l = (t * 24 + h) * 48 + w;
  const float2 x = *(const float2*)(k + ((size_t)l * HEADS + head) * HDIM + 2 * p);
  float y0, y1;
  rope_rotate(pos, p, x.x, x.y, y0, y1);
  union { _Float16 h2[2]; unsigned u; } pk;
  pk.h2[0] = (_Float16)y0;
  pk.h2[1] = (_Float16)y1;
  *(unsigned*)(Kh + (((size_t)(head * KVBLK + kb) * BS + row) * HDIM + 2 * p)) = pk.u;
}

// V preprocess: glue + tile + transpose -> Vh[head][kb][e][row] f16 (dim-major)
__global__ __launch_bounds__(256) void gather_v_kernel(const float* __restrict__ v,
                                                       _Float16* __restrict__ Vh) {
  int gid = blockIdx.x * 256 + threadIdx.x;     // HEADS*KVBLK*HDIM*(BS/2)
  int rp = gid & 127;              // row pair
  int e = (gid >> 7) & 127;
  int kb = (gid >> 14) % KVBLK;
  int head = (gid >> 14) / KVBLK;
  int kt = kb / 40, kh = (kb / 8) % 5, kw = kb % 8;
  union { _Float16 h2[2]; unsigned u; } pk;
#pragma unroll
  for (int s = 0; s < 2; s++) {
    int row = rp * 2 + s;
    int tt = row >> 6, th = (row >> 3) & 7, tw = row & 7;
    int t = kt * 4 + tt;
    int h = (kh * 8 + th + 16) % 24;
    int w = (kw * 8 + tw + 40) % 48;
    int l = (t * 24 + h) * 48 + w;
    pk.h2[s] = (_Float16)v[((size_t)l * HEADS + head) * HDIM + e];
  }
  *(unsigned*)(Vh + (((size_t)(head * KVBLK + kb) * HDIM + e) * BS + rp * 2)) = pk.u;
}

// ---------------------------------------------------------------------------
// Async staging: one KV block (64KB K + 64KB V) into an LDS buffer via
// GLOBAL_LOAD_ASYNC_TO_LDS_B128. 16 async b128 per thread => ASYNCcnt += 16
// per wave per buffer. Async loads complete in order, so waiting ASYNCcnt<=16
// guarantees the previous buffer is resident while the next streams in.
// ---------------------------------------------------------------------------
__device__ __forceinline__ void async_stage(const _Float16* gK, const _Float16* gV,
                                            unsigned ldsK, unsigned ldsV, int tid) {
#pragma unroll
  for (int it = 0; it < 8; it++) {
    const int byteoff = (it * 512 + tid) * 16;   // 4096 x 16B per buffer
    unsigned long long gk = (unsigned long long)(const char*)gK + byteoff;
    unsigned long long gv = (unsigned long long)(const char*)gV + byteoff;
    unsigned lk = ldsK + byteoff;
    unsigned lv = ldsV + byteoff;
    asm volatile("global_load_async_to_lds_b128 %0, %1, off"
                 :: "v"(lk), "v"(gk) : "memory");
    asm volatile("global_load_async_to_lds_b128 %0, %1, off"
                 :: "v"(lv), "v"(gv) : "memory");
  }
}

// ---------------------------------------------------------------------------
// Attention: one workgroup per (qb, head). 512 threads = 16 waves; wave w owns
// query rows [16w, 16w+16). Flash-attention over 27 KV blocks, double-buffered
// async K/V staging in LDS, all GEMMs via v_wmma_f32_16x16x32_f16.
// ---------------------------------------------------------------------------
__global__ __launch_bounds__(512) void sta_attn_kernel(const _Float16* __restrict__ Qh,
                                                       const _Float16* __restrict__ Kh,
                                                       const _Float16* __restrict__ Vh,
                                                       float* __restrict__ out) {
  extern __shared__ char smem_raw[];
  _Float16* sK = (_Float16*)smem_raw;       // 2 x [256 tok][128 e]  128 KB
  _Float16* sV = sK + 2 * BS * HDIM;        // 2 x [128 e][256 tok]  128 KB
  _Float16* sP = sV + 2 * HDIM * BS;        // 16 waves * 16*64       32 KB

  const int tid = threadIdx.x;
  const int wave = tid >> 5;
  const int lane = tid & 31;
  const int l15 = lane & 15;
  const int hi = lane >> 4;

  const int qb = blockIdx.x % QBLK;
  const int head = blockIdx.x / QBLK;
  const int bt = qb / 18, bh = (qb / 6) % 3, bw = qb % 6;

  const unsigned ldsK0 = (unsigned)(unsigned long long)sK;  // LDS byte offsets
  const unsigned ldsV0 = (unsigned)(unsigned long long)sV;
  const _Float16* KhH = Kh + (size_t)head * KVBLK * (BS * HDIM);
  const _Float16* VhH = Vh + (size_t)head * KVBLK * (BS * HDIM);

  // Q fragments (A-layout 16x32 f16): lane<16 row=l15 cols c*32+[0..15]
  //                                   lane>=16         cols c*32+[16..31]
  v16h aQ[4];
  {
    const _Float16* qbase = Qh + (size_t)(head * QBLK + qb) * BS * HDIM;
    const int qrow = wave * 16 + l15;
#pragma unroll
    for (int c = 0; c < 4; c++) {
      const _Float16* pQ = qbase + qrow * HDIM + c * 32 + hi * 16;
      union { v16h h; uint4 u[2]; } af;
      af.u[0] = *(const uint4*)pQ;
      af.u[1] = *(const uint4*)(pQ + 8);
      aQ[c] = af.h;
    }
  }

  v8f O[8];
  float m[8], lsum[8];
#pragma unroll
  for (int d = 0; d < 8; d++)
#pragma unroll
    for (int j = 0; j < 8; j++) O[d][j] = 0.0f;
#pragma unroll
  for (int j = 0; j < 8; j++) { m[j] = -3.402823466e38f; lsum[j] = 0.0f; }

  const float SC = 0.08838834764831845f * 1.4426950408889634f;  // 1/sqrt(128)*log2(e)
  _Float16* sPw = sP + wave * (16 * 64);

  // prologue: stage block 0 into buffer 0
  {
    const int kb0 = bh * 8 + bw;       // ib=0 -> dt=dh=dw=0
    async_stage(KhH + (size_t)kb0 * (BS * HDIM), VhH + (size_t)kb0 * (BS * HDIM),
                ldsK0, ldsV0, tid);
  }

  for (int ib = 0; ib < 27; ib++) {
    const int cur = ib & 1;
    // issue next block into the other buffer (safe: trailing barrier of the
    // previous iteration guarantees all waves finished reading it)
    if (ib + 1 < 27) {
      const int jb = ib + 1;
      const int kbn = (jb / 9) * 40 + (bh + (jb / 3) % 3) * 8 + (bw + jb % 3);
      async_stage(KhH + (size_t)kbn * (BS * HDIM), VhH + (size_t)kbn * (BS * HDIM),
                  ldsK0 + (1 - cur) * (BS * HDIM * 2), ldsV0 + (1 - cur) * (BS * HDIM * 2),
                  tid);
      asm volatile("s_wait_asynccnt 0x10" ::: "memory");  // current buffer done
    } else {
      asm volatile("s_wait_asynccnt 0x0" ::: "memory");
    }
    __syncthreads();

    const _Float16* sKc = sK + cur * (BS * HDIM);
    const _Float16* sVc = sV + cur * (HDIM * BS);

    // ---- 4 chunks of 64 KV tokens ----
    for (int ch = 0; ch < 4; ch++) {
      v8f S[4];
#pragma unroll
      for (int t = 0; t < 4; t++)
#pragma unroll
        for (int j = 0; j < 8; j++) S[t][j] = 0.0f;

      // S = Q * K^T ; batch 8 ds_load_b128 then 4 wmma per k-chunk
#pragma unroll
      for (int cK = 0; cK < 4; cK++) {
        union { v16h h; uint4 u[2]; } bf[4];
#pragma unroll
        for (int t = 0; t < 4; t++) {
          const _Float16* pB = sKc + (ch * 64 + t * 16 + l15) * HDIM + cK * 32 + hi * 16;
          bf[t].u[0] = *(const uint4*)pB;
          bf[t].u[1] = *(const uint4*)(pB + 8);
        }
#pragma unroll
        for (int t = 0; t < 4; t++)
          S[t] = __builtin_amdgcn_wmma_f32_16x16x32_f16(
              false, aQ[cK], false, bf[t].h, (short)0, S[t], false, false);
      }

      // ---- online softmax (row stats in C-layout: reg j = row j+8*hi) ----
      float alpha[8];
#pragma unroll
      for (int j = 0; j < 8; j++) {
        float x = fmaxf(fmaxf(S[0][j], S[1][j]), fmaxf(S[2][j], S[3][j]));
#pragma unroll
        for (int off = 1; off < 16; off <<= 1)
          x = fmaxf(x, __shfl_xor(x, off, 32));
        float mn = fmaxf(m[j], x);
        alpha[j] = exp2f((m[j] - mn) * SC);
        m[j] = mn;
      }
#pragma unroll
      for (int j = 0; j < 8; j++) {
        float rs = 0.0f;
#pragma unroll
        for (int t = 0; t < 4; t++) {
          float p = exp2f((S[t][j] - m[j]) * SC);
          S[t][j] = p;
          rs += p;
        }
#pragma unroll
        for (int off = 1; off < 16; off <<= 1)
          rs += __shfl_xor(rs, off, 32);
        lsum[j] = lsum[j] * alpha[j] + rs;
      }
#pragma unroll
      for (int d = 0; d < 8; d++)
#pragma unroll
        for (int j = 0; j < 8; j++) O[d][j] *= alpha[j];

      // ---- transpose P: C-layout f32 -> A-layout f16 via per-wave LDS ----
#pragma unroll
      for (int t = 0; t < 4; t++)
#pragma unroll
        for (int j = 0; j < 8; j++)
          sPw[(j + hi * 8) * 64 + t * 16 + l15] = (_Float16)S[t][j];
      asm volatile("s_wait_dscnt 0" ::: "memory");

      v16h aP[2];
#pragma unroll
      for (int c2 = 0; c2 < 2; c2++) {
        const _Float16* pA = sPw + l15 * 64 + c2 * 32 + hi * 16;
        union { v16h h; uint4 u[2]; } af;
        af.u[0] = *(const uint4*)pA;
        af.u[1] = *(const uint4*)(pA + 8);
        aP[c2] = af.h;
      }

      // ---- O += P * V (V dim-major: lane holds 16 contiguous tokens) ----
#pragma unroll
      for (int c2 = 0; c2 < 2; c2++) {
        const int tok0 = ch * 64 + c2 * 32 + hi * 16;
#pragma unroll
        for (int dd = 0; dd < 8; dd += 2) {
          union { v16h h; uint4 u[2]; } bf0, bf1;
          const _Float16* pB0 = sVc + (dd * 16 + l15) * BS + tok0;
          const _Float16* pB1 = sVc + ((dd + 1) * 16 + l15) * BS + tok0;
          bf0.u[0] = *(const uint4*)pB0;
          bf0.u[1] = *(const uint4*)(pB0 + 8);
          bf1.u[0] = *(const uint4*)pB1;
          bf1.u[1] = *(const uint4*)(pB1 + 8);
          O[dd] = __builtin_amdgcn_wmma_f32_16x16x32_f16(
              false, aP[c2], false, bf0.h, (short)0, O[dd], false, false);
          O[dd + 1] = __builtin_amdgcn_wmma_f32_16x16x32_f16(
              false, aP[c2], false, bf1.h, (short)0, O[dd + 1], false, false);
        }
      }
    }
    __syncthreads();  // all waves done with buffer `cur` before it is refilled
  }

  // ---- epilogue: O/l, untile, write (B,L,H,E) ----
#pragma unroll
  for (int j = 0; j < 8; j++) {
    const float inv = 1.0f / lsum[j];
    const int r = wave * 16 + j + hi * 8;
    const int tt = r >> 6, th = (r >> 3) & 7, tw = r & 7;
    const int t = bt * 4 + tt, h = bh * 8 + th, w = bw * 8 + tw;
    const size_t l = ((size_t)t * 24 + h) * 48 + w;
    float* ob = out + (l * HEADS + head) * HDIM + l15;
#pragma unroll
    for (int d = 0; d < 8; d++) ob[d * 16] = O[d][j] * inv;
  }
}

extern "C" void kernel_launch(void* const* d_in, const int* in_sizes, int n_in,
                              void* d_out, int out_size, void* d_ws, size_t ws_size,
                              hipStream_t stream) {
  const float* q = (const float*)d_in[0];
  const float* k = (const float*)d_in[1];
  const float* v = (const float*)d_in[2];
  float* out = (float*)d_out;

  const size_t QH_ELEMS = (size_t)HEADS * QBLK * BS * HDIM;    // 42,467,328
  const size_t KH_ELEMS = (size_t)HEADS * KVBLK * BS * HDIM;   // 94,371,840
  _Float16* Qh = (_Float16*)d_ws;
  _Float16* Kh = Qh + QH_ELEMS;
  _Float16* Vh = Kh + KH_ELEMS;

  {
    int total = QBLK * BS * HEADS * 64;     // 21,233,664
    rope_q_kernel<<<total / 256, 256, 0, stream>>>(q, Qh);
  }
  {
    int total = KVBLK * BS * HEADS * 64;    // 47,185,920
    rope_k_kernel<<<total / 256, 256, 0, stream>>>(k, Kh);
  }
  {
    int total = HEADS * KVBLK * HDIM * (BS / 2);  // 47,185,920
    gather_v_kernel<<<total / 256, 256, 0, stream>>>(v, Vh);
  }
  {
    const size_t smem = (size_t)(4 * BS * HDIM + 16 * 16 * 64) * sizeof(_Float16); // 288 KB
    sta_attn_kernel<<<QBLK * HEADS, 512, smem, stream>>>(Qh, Kh, Vh, out);
  }
}
